// Lfm2MoeSparseMoeBlock_2113123909695
// MI455X (gfx1250) — matrix operand order repaired and checked by reading, not compiled, not run on hardware
//
#include <hip/hip_runtime.h>
#include <hip/hip_bf16.h>

// ---------------------------------------------------------------------------
// LFM2 MoE sparse block for MI455X (gfx1250, wave32, WMMA).
// Roofline: 38.7 GFLOP useful work, 302 MB fp32 weights -> ~13us HBM floor.
// bf16 WMMA (16x16x32, f32 accum) balances compute against that floor;
// fp32 WMMA (K=4) would be ~8x compute-bound.
// L2 is the second-order limit: each workgroup re-reads its expert's weights
// (9 MB) from the 192MB L2, so TILE_M=64 (4 WMMA M-tiles per B fragment)
// halves L2 traffic vs TILE_M=32 (~1.2 GB total). LDS = 232 KB/WG (<320 KB).
// ---------------------------------------------------------------------------

#define E_ 32
#define K_ 4
#define H_ 1024
#define I_ 768
#define T_ 2048

#define TILE_M      64
#define MT_         (TILE_M / 16)   // 4 WMMA M-tiles
#define SA_STRIDE   (H_ + 8)        // 1032 bf16 -> 2064B row stride (16B align)
#define SACT_STRIDE (I_ + 8)        // 776  bf16 -> 1552B row stride (16B align)

typedef __attribute__((ext_vector_type(16))) __bf16 bf16x16;
typedef __attribute__((ext_vector_type(8)))  __bf16 bf16x8;
typedef __attribute__((ext_vector_type(4)))  __bf16 bf16x4;
typedef __attribute__((ext_vector_type(8)))  float  f32x8;

__device__ __forceinline__ f32x8 zero8() {
    f32x8 z;
#pragma unroll
    for (int i = 0; i < 8; ++i) z[i] = 0.0f;
    return z;
}

// fast sigmoid: v_exp_f32 + v_rcp_f32 (no IEEE divide chain)
__device__ __forceinline__ float sigmoidf_fast(float v) {
    return __builtin_amdgcn_rcpf(1.0f + __expf(-v));
}

// A-fragment (16x32 bf16) from LDS:
// lanes 0-15 row=l:    K [kb..kb+15]    (ks = kb)
// lanes 16-31 row=l-16: K [kb+16..kb+31] (ks = kb+16)
__device__ __forceinline__ bf16x16 load_a_frag(const __bf16* rowp, int ks) {
    bf16x8 lo = *(const bf16x8*)(rowp + ks);
    bf16x8 hi = *(const bf16x8*)(rowp + ks + 8);
    bf16x16 A;
#pragma unroll
    for (int i = 0; i < 8; ++i) { A[i] = lo[i]; A[8 + i] = hi[i]; }
    return A;
}

// B-fragment (32x16 bf16): lanes 0-15 col=l hold K kb..kb+15,
// lanes 16-31 col=l-16 hold K kb+16..kb+31. Gather fp32, convert inline.
__device__ __forceinline__ bf16x16 load_b_frag(const float* __restrict__ p, int ld) {
    bf16x16 B;
#pragma unroll
    for (int i = 0; i < 16; ++i) B[i] = (__bf16)p[(size_t)i * ld];
    return B;
}

// ---------------------------------------------------------------------------
// Kernel 1: router. One wave32 per token; lane e owns expert e.
// ---------------------------------------------------------------------------
__global__ __launch_bounds__(256) void moe_router(
    const float* __restrict__ x, const float* __restrict__ gw,
    const float* __restrict__ ebias,
    int* __restrict__ topk_ids, float* __restrict__ topk_w,
    int* __restrict__ ecount)
{
    const int wave = threadIdx.x >> 5;
    const int lane = threadIdx.x & 31;
    const int t = blockIdx.x * 8 + wave;
    if (t >= T_) return;

    const float4* xp = (const float4*)(x + (size_t)t * H_);
    const float4* gp = (const float4*)(gw + (size_t)lane * H_);
    float acc = 0.0f;
#pragma unroll 4
    for (int i = 0; i < H_ / 4; ++i) {
        float4 a = xp[i];
        float4 b = gp[i];
        acc += a.x * b.x + a.y * b.y + a.z * b.z + a.w * b.w;
    }
    const float score  = sigmoidf_fast(acc);
    float       biased = score + ebias[lane];

    int   sel_id[K_];
    float sel_w[K_];
#pragma unroll
    for (int k = 0; k < K_; ++k) {
        float v = biased;
        int   idx = lane;
        // wave32 argmax reduction (lowest index wins ties, like lax.top_k)
#pragma unroll
        for (int off = 16; off >= 1; off >>= 1) {
            float ov = __shfl_xor(v, off, 32);
            int   oi = __shfl_xor(idx, off, 32);
            if (ov > v || (ov == v && oi < idx)) { v = ov; idx = oi; }
        }
        sel_id[k] = idx;
        sel_w[k]  = __shfl(score, idx, 32);   // raw sigmoid score of winner
        if (lane == idx) biased = -1e30f;     // knock out for next round
    }
    const float inv = __builtin_amdgcn_rcpf(sel_w[0] + sel_w[1] + sel_w[2] + sel_w[3]);
    if (lane < K_) {
        topk_ids[t * K_ + lane] = sel_id[lane];
        topk_w[t * K_ + lane]   = sel_w[lane] * inv;
        atomicAdd(&ecount[sel_id[lane]], 1);
    }
}

// ---------------------------------------------------------------------------
// Kernel 2: exclusive scan of expert counts (E=32, serial is fine).
// ---------------------------------------------------------------------------
__global__ void moe_scan(const int* __restrict__ ecount, int* __restrict__ eoff)
{
    if (threadIdx.x == 0 && blockIdx.x == 0) {
        int run = 0;
        for (int e = 0; e < E_; ++e) { eoff[e] = run; run += ecount[e]; }
        eoff[E_] = run;
    }
}

// ---------------------------------------------------------------------------
// Kernel 3: scatter tokens into per-expert contiguous lists.
// ---------------------------------------------------------------------------
__global__ __launch_bounds__(256) void moe_scatter(
    const int* __restrict__ topk_ids, const float* __restrict__ topk_w,
    const int* __restrict__ eoff, int* __restrict__ efill,
    int* __restrict__ tok_list, float* __restrict__ wt_list)
{
    const int t = blockIdx.x * 256 + threadIdx.x;
    if (t >= T_) return;
#pragma unroll
    for (int k = 0; k < K_; ++k) {
        const int e   = topk_ids[t * K_ + k];
        const int pos = atomicAdd(&efill[e], 1);
        const int idx = eoff[e] + pos;
        tok_list[idx] = t;
        wt_list[idx]  = topk_w[t * K_ + k];
    }
}

// ---------------------------------------------------------------------------
// Kernel 4: fused expert SwiGLU + down-proj, V_WMMA_F32_16X16X32_BF16.
// Block = 8 wave32s, one (expert, 64-token tile).
//   Phase 0: gather X tile -> bf16 LDS
//   Phase 1: G=X@W1, U=X@W3, act = silu(G)*U*route_w -> bf16 LDS
//   Phase 2: out += act @ W2, f32 atomic scatter-add per token row
// ---------------------------------------------------------------------------
__global__ __launch_bounds__(256) void moe_expert(
    const float* __restrict__ x,
    const float* __restrict__ w1, const float* __restrict__ w3,
    const float* __restrict__ w2,
    const int* __restrict__ eoff, const int* __restrict__ tok_list,
    const float* __restrict__ wt_list, float* __restrict__ out)
{
    __shared__ __bf16 sA[TILE_M * SA_STRIDE];      // 132096 B
    __shared__ __bf16 sAct[TILE_M * SACT_STRIDE];  //  99328 B
    __shared__ float  sW[TILE_M];
    __shared__ int    sTok[TILE_M];

    const int e     = blockIdx.y;
    const int start = eoff[e];
    const int n_e   = eoff[e + 1] - start;
    const int mbase = blockIdx.x * TILE_M;
    if (mbase >= n_e) return;                       // empty tile -> fast exit
    const int n_tok = (n_e - mbase < TILE_M) ? (n_e - mbase) : TILE_M;

    const int tid  = threadIdx.x;
    const int wave = tid >> 5;
    const int lane = tid & 31;
    const int ln15 = lane & 15;
    const int hi   = lane >> 4;                     // 0 | 1: lane half

    // ---- token metadata
    if (tid < TILE_M) {
        if (tid < n_tok) {
            sTok[tid] = tok_list[start + mbase + tid];
            sW[tid]   = wt_list[start + mbase + tid];
        } else {
            sTok[tid] = -1;
            sW[tid]   = 0.0f;
        }
    }

    // ---- Phase 0: gather X tile, fp32 -> bf16, into LDS (4 thr/row)
    {
        const int r  = tid >> 2;                    // 0..63
        const int c0 = tid & 3;
        const int rr = (r < n_tok) ? r : 0;         // clamp: safe address
        const int tok = tok_list[start + mbase + rr];
        const float4* xrow = (const float4*)(x + (size_t)tok * H_);
        for (int c4 = c0; c4 < H_ / 4; c4 += 4) {
            float4 v = xrow[c4];
            if (r >= n_tok) { v.x = 0.f; v.y = 0.f; v.z = 0.f; v.w = 0.f; }
            bf16x4 b;
            b[0] = (__bf16)v.x; b[1] = (__bf16)v.y;
            b[2] = (__bf16)v.z; b[3] = (__bf16)v.w;
            *(bf16x4*)(&sA[r * SA_STRIDE + c4 * 4]) = b;
        }
    }
    __syncthreads();

    const int ksel = hi ? 16 : 0;       // A-frag sub-chunk select
    const int koff = hi ? 16 : 0;       // B-frag K base select

    // ---- Phase 1: up + gate projections, SwiGLU epilogue
    const float* w1e = w1 + (size_t)e * H_ * I_;
    const float* w3e = w3 + (size_t)e * H_ * I_;

    for (int nt = wave; nt < I_ / 16; nt += 8) {
        const int cb = nt * 16 + ln15;              // my output column
        f32x8 g[MT_], u[MT_];
#pragma unroll
        for (int mt = 0; mt < MT_; ++mt) { g[mt] = zero8(); u[mt] = zero8(); }

        for (int kb = 0; kb < H_; kb += 32) {
            const float* pg = w1e + (size_t)(kb + koff) * I_ + cb;
            const float* pu = w3e + (size_t)(kb + koff) * I_ + cb;
            __builtin_prefetch(pg + 32 * I_, 0, 0);     // next k-step rows
            __builtin_prefetch(pu + 32 * I_, 0, 0);
            bf16x16 Bg = load_b_frag(pg, I_);
            bf16x16 Bu = load_b_frag(pu, I_);
#pragma unroll
            for (int mt = 0; mt < MT_; ++mt) {
                bf16x16 A = load_a_frag(&sA[(mt * 16 + ln15) * SA_STRIDE], kb + ksel);
                g[mt] = __builtin_amdgcn_wmma_f32_16x16x32_bf16(false, A, false, Bg, (short)0, g[mt], false, false);
                u[mt] = __builtin_amdgcn_wmma_f32_16x16x32_bf16(false, A, false, Bu, (short)0, u[mt], false, false);
            }
        }
        // epilogue: act = silu(g) * u * route_w -> bf16 LDS
#pragma unroll
        for (int mt = 0; mt < MT_; ++mt) {
#pragma unroll
            for (int r = 0; r < 8; ++r) {
                const int row = mt * 16 + r + hi * 8;
                const float gg = g[mt][r];
                const float av = gg * sigmoidf_fast(gg) * u[mt][r] * sW[row];
                sAct[row * SACT_STRIDE + nt * 16 + ln15] = (__bf16)av;
            }
        }
    }
    __syncthreads();

    // ---- Phase 2: down projection, atomic scatter into out
    const float* w2e = w2 + (size_t)e * I_ * H_;

    for (int nt = wave; nt < H_ / 16; nt += 8) {
        const int cb = nt * 16 + ln15;
        f32x8 o[MT_];
#pragma unroll
        for (int mt = 0; mt < MT_; ++mt) o[mt] = zero8();

        for (int kb = 0; kb < I_; kb += 32) {
            const float* p2 = w2e + (size_t)(kb + koff) * H_ + cb;
            __builtin_prefetch(p2 + 32 * H_, 0, 0);
            bf16x16 B2 = load_b_frag(p2, H_);
#pragma unroll
            for (int mt = 0; mt < MT_; ++mt) {
                bf16x16 A = load_a_frag(&sAct[(mt * 16 + ln15) * SACT_STRIDE], kb + ksel);
                o[mt] = __builtin_amdgcn_wmma_f32_16x16x32_bf16(false, A, false, B2, (short)0, o[mt], false, false);
            }
        }
#pragma unroll
        for (int mt = 0; mt < MT_; ++mt) {
#pragma unroll
            for (int r = 0; r < 8; ++r) {
                const int row = mt * 16 + r + hi * 8;
                const int tk = sTok[row];
                if (tk >= 0) atomicAdd(&out[(size_t)tk * H_ + cb], o[mt][r]);
            }
        }
    }
}

// ---------------------------------------------------------------------------
// Host launcher (graph-capture safe: only async stream ops).
// Workspace layout: [counts | fill | off | topk_ids | topk_w | tok_list |
//                    wt_list]  (~132 KB total)
// ---------------------------------------------------------------------------
extern "C" void kernel_launch(void* const* d_in, const int* in_sizes, int n_in,
                              void* d_out, int out_size, void* d_ws, size_t ws_size,
                              hipStream_t stream) {
    const float* x     = (const float*)d_in[0];
    const float* gw    = (const float*)d_in[1];
    const float* w1    = (const float*)d_in[2];
    const float* w3    = (const float*)d_in[3];
    const float* w2    = (const float*)d_in[4];
    const float* ebias = (const float*)d_in[5];
    float* out = (float*)d_out;

    char* ws = (char*)d_ws;
    int*   ecount   = (int*)(ws + 0);
    int*   efill    = (int*)(ws + 256);
    int*   eoff     = (int*)(ws + 512);
    int*   topk_ids = (int*)(ws + 1024);
    float* topk_w   = (float*)(ws + 1024 + 1 * (size_t)T_ * K_ * 4);
    int*   tok_list = (int*)(ws + 1024 + 2 * (size_t)T_ * K_ * 4);
    float* wt_list  = (float*)(ws + 1024 + 3 * (size_t)T_ * K_ * 4);

    hipMemsetAsync(ws, 0, 1024, stream);                                // counters
    hipMemsetAsync(out, 0, (size_t)T_ * H_ * sizeof(float), stream);    // accum target

    moe_router<<<T_ / 8, 256, 0, stream>>>(x, gw, ebias, topk_ids, topk_w, ecount);
    moe_scan<<<1, 32, 0, stream>>>(ecount, eoff);
    moe_scatter<<<T_ / 256, 256, 0, stream>>>(topk_ids, topk_w, eoff, efill,
                                              tok_list, wt_list);
    dim3 grid(T_ / TILE_M, E_);   // 32 tiles covers worst-case expert load
    moe_expert<<<grid, 256, 0, stream>>>(x, w1, w3, w2, eoff, tok_list, wt_list, out);
}
